// MyLinearSlct_75015898792455
// MI455X (gfx1250) — matrix operand order repaired and checked by reading, not compiled, not run on hardware
//
#include <hip/hip_runtime.h>

// MoE top-1 linear: y[b,:] = relu(W[argmax(slct[b])] @ x[b] + bias[argmax(slct[b])])
// B=8192, NIN=NHID=1024, E=8.
// Compute-bound (AI ~180 FLOP/B) -> split-bf16 (hi/lo) x3 on V_WMMA_F32_16X16X32_BF16:
// ~2.7x fewer matrix-pipe issues than the f32 16x16x4 path at near-fp32 accuracy.

typedef __bf16 bf16_t;
typedef bf16_t v8bf  __attribute__((ext_vector_type(8)));
typedef bf16_t v16bf __attribute__((ext_vector_type(16)));
typedef float  v8f   __attribute__((ext_vector_type(8)));

#define B_ROWS 8192
#define NIN    1024
#define NHID   1024
#define NSLCT  8

#define TM 128           // M tile (rows of x, gathered per expert)
#define TN 64            // N tile (output columns / W rows)
#define KC 32            // K chunk = one bf16 WMMA K
#define LDK (KC + 8)     // 40 bf16 = 80B row stride: 16B-aligned, bank-conflict free
#define THREADS 256      // 8 wave32s: 4 along M (32 rows) x 2 along N (32 cols)

// d_ws int layout:
//  [0..7]   counts per expert
//  [8..15]  segment offsets (exclusive prefix sum)
//  [16..23] scatter cursors
//  [24..24+B_ROWS) perm: row indices grouped by expert
#define WS_COUNTS  0
#define WS_OFFSETS 8
#define WS_CURSOR  16
#define WS_PERM    24

__global__ void init_ws_kernel(int* ws) {
  int t = threadIdx.x;
  if (t < 24) ws[t] = 0;
}

__global__ void argmax_count_kernel(const float* __restrict__ slct,
                                    int* __restrict__ ws) {
  int b = blockIdx.x * blockDim.x + threadIdx.x;
  if (b >= B_ROWS) return;
  const float* s = slct + (size_t)b * NSLCT;
  float best = s[0];
  int bi = 0;
#pragma unroll
  for (int j = 1; j < NSLCT; ++j) {
    float v = s[j];
    if (v > best) { best = v; bi = j; }   // first-max tie-break, like jnp.argmax
  }
  atomicAdd(&ws[WS_COUNTS + bi], 1);
}

__global__ void scan_offsets_kernel(int* ws) {
  if (threadIdx.x == 0 && blockIdx.x == 0) {
    int run = 0;
    for (int e = 0; e < NSLCT; ++e) {
      ws[WS_OFFSETS + e] = run;
      run += ws[WS_COUNTS + e];
    }
  }
}

__global__ void scatter_rows_kernel(const float* __restrict__ slct,
                                    int* __restrict__ ws) {
  int b = blockIdx.x * blockDim.x + threadIdx.x;
  if (b >= B_ROWS) return;
  const float* s = slct + (size_t)b * NSLCT;
  float best = s[0];
  int bi = 0;
#pragma unroll
  for (int j = 1; j < NSLCT; ++j) {
    float v = s[j];
    if (v > best) { best = v; bi = j; }
  }
  int pos = atomicAdd(&ws[WS_CURSOR + bi], 1);
  ws[WS_PERM + ws[WS_OFFSETS + bi] + pos] = b;
}

// Grouped GEMM over per-expert row segments, split-bf16 x3 accumulation.
__global__ __launch_bounds__(THREADS)
void moe_gemm_kernel(const float* __restrict__ x,
                     const float* __restrict__ W,
                     const float* __restrict__ bias,
                     const int*   __restrict__ ws,
                     float* __restrict__ out) {
  const int e      = blockIdx.z;
  const int segLen = ws[WS_COUNTS + e];
  const int mBase  = blockIdx.y * TM;
  if (mBase >= segLen) return;                  // uniform early exit
  const int segStart = ws[WS_OFFSETS + e];
  const int nBase    = blockIdx.x * TN;
  const int* perm    = ws + WS_PERM;

  __shared__ bf16_t sAh[TM][LDK];               // x tile hi plane, [m][k]
  __shared__ bf16_t sAl[TM][LDK];               // x tile lo plane
  __shared__ bf16_t sBh[TN][LDK];               // W tile hi plane, [n][k] (natural)
  __shared__ bf16_t sBl[TN][LDK];               // W tile lo plane
  __shared__ int    sRow[TM];                   // original row index (-1 = pad)

  const int t = threadIdx.x;
  if (t < TM) {
    int gm = mBase + t;
    sRow[t] = (gm < segLen) ? perm[segStart + gm] : -1;
  }
  __syncthreads();

  const int lane  = t & 31;
  const int wid   = t >> 5;
  const int waveM = (wid & 3) * 32;
  const int waveN = (wid >> 2) * 32;
  const int lm    = lane & 15;
  const int kbA   = (lane >> 4) * 8;            // A: lanes 16-31 carry K+8 / K+24
  const int kbB   = (lane >> 4) * 8;            // B: lanes 16-31 carry K+8 / K+24

  // Staging assignments: A -> 16 floats of one row per thread; B -> 8 floats.
  const int arow = t >> 1;                      // 0..127
  const int ak0  = (t & 1) * 16;                // 0 or 16
  const int brow = t >> 2;                      // 0..63
  const int bk0  = (t & 3) * 8;                 // 0,8,16,24
  const int rowA = sRow[arow];
  const float* Wb = W + (size_t)e * NHID * NIN + (size_t)(nBase + brow) * NIN;

  v8f acc[2][2] = {};

  for (int kc = 0; kc < NIN; kc += KC) {
    // ---- global loads (f32), before barrier ----
    float av[16];
    if (rowA >= 0) {
      const float* xr = x + (size_t)rowA * NIN + kc + ak0;
#pragma unroll
      for (int j = 0; j < 4; ++j) {
        float4 f = *(const float4*)(xr + j * 4);
        av[j * 4 + 0] = f.x; av[j * 4 + 1] = f.y;
        av[j * 4 + 2] = f.z; av[j * 4 + 3] = f.w;
      }
    } else {
#pragma unroll
      for (int j = 0; j < 16; ++j) av[j] = 0.f;
    }
    float bv[8];
#pragma unroll
    for (int j = 0; j < 2; ++j) {
      float4 f = *(const float4*)(Wb + kc + bk0 + j * 4);
      bv[j * 4 + 0] = f.x; bv[j * 4 + 1] = f.y;
      bv[j * 4 + 2] = f.z; bv[j * 4 + 3] = f.w;
    }

    // ---- split f32 -> bf16 hi + bf16 lo, pack to 16B vectors ----
    v8bf ah0, ah1, al0, al1;
#pragma unroll
    for (int j = 0; j < 8; ++j) {
      bf16_t h = (bf16_t)av[j];
      ah0[j] = h;
      al0[j] = (bf16_t)(av[j] - (float)h);
    }
#pragma unroll
    for (int j = 0; j < 8; ++j) {
      bf16_t h = (bf16_t)av[8 + j];
      ah1[j] = h;
      al1[j] = (bf16_t)(av[8 + j] - (float)h);
    }
    v8bf bh0, bl0;
#pragma unroll
    for (int j = 0; j < 8; ++j) {
      bf16_t h = (bf16_t)bv[j];
      bh0[j] = h;
      bl0[j] = (bf16_t)(bv[j] - (float)h);
    }

    __syncthreads();                            // prior iteration done reading LDS
    *(v8bf*)&sAh[arow][ak0]     = ah0;
    *(v8bf*)&sAh[arow][ak0 + 8] = ah1;
    *(v8bf*)&sAl[arow][ak0]     = al0;
    *(v8bf*)&sAl[arow][ak0 + 8] = al1;
    *(v8bf*)&sBh[brow][bk0]     = bh0;
    *(v8bf*)&sBl[brow][bk0]     = bl0;
    __syncthreads();

    // ---- load fragments (ISA 7.12.2 16-bit layouts) ----
    v16bf ah[2], al[2], bh[2], bl[2];
#pragma unroll
    for (int mi = 0; mi < 2; ++mi) {
      const int m = waveM + mi * 16 + lm;
      ((v8bf*)&ah[mi])[0] = *(const v8bf*)&sAh[m][kbA];
      ((v8bf*)&ah[mi])[1] = *(const v8bf*)&sAh[m][16 + kbA];
      ((v8bf*)&al[mi])[0] = *(const v8bf*)&sAl[m][kbA];
      ((v8bf*)&al[mi])[1] = *(const v8bf*)&sAl[m][16 + kbA];
    }
#pragma unroll
    for (int ni = 0; ni < 2; ++ni) {
      const int n = waveN + ni * 16 + lm;
      ((v8bf*)&bh[ni])[0] = *(const v8bf*)&sBh[n][kbB];
      ((v8bf*)&bh[ni])[1] = *(const v8bf*)&sBh[n][16 + kbB];
      ((v8bf*)&bl[ni])[0] = *(const v8bf*)&sBl[n][kbB];
      ((v8bf*)&bl[ni])[1] = *(const v8bf*)&sBl[n][16 + kbB];
    }

    // ---- 3-term split-bf16 accumulation: hi*hi + hi*lo + lo*hi ----
#pragma unroll
    for (int mi = 0; mi < 2; ++mi) {
#pragma unroll
      for (int ni = 0; ni < 2; ++ni) {
        acc[mi][ni] = __builtin_amdgcn_wmma_f32_16x16x32_bf16(
            false, ah[mi], false, bh[ni], (short)0, acc[mi][ni], false, false);
        acc[mi][ni] = __builtin_amdgcn_wmma_f32_16x16x32_bf16(
            false, ah[mi], false, bl[ni], (short)0, acc[mi][ni], false, false);
        acc[mi][ni] = __builtin_amdgcn_wmma_f32_16x16x32_bf16(
            false, al[mi], false, bh[ni], (short)0, acc[mi][ni], false, false);
      }
    }
  }

  // ---- epilogue: bias + relu, scatter back to original rows ----
#pragma unroll
  for (int mi = 0; mi < 2; ++mi) {
#pragma unroll
    for (int ni = 0; ni < 2; ++ni) {
      const int n  = waveN + ni * 16 + lm;
      const float bcol = bias[(size_t)e * NHID + nBase + n];
#pragma unroll
      for (int v = 0; v < 8; ++v) {
        const int m = waveM + mi * 16 + (lane >> 4) * 8 + v;  // C layout 7.12.2
        const int row = sRow[m];
        if (row >= 0) {
          float r = acc[mi][ni][v] + bcol;
          out[(size_t)row * NHID + nBase + n] = r > 0.f ? r : 0.f;
        }
      }
    }
  }
}

extern "C" void kernel_launch(void* const* d_in, const int* in_sizes, int n_in,
                              void* d_out, int out_size, void* d_ws, size_t ws_size,
                              hipStream_t stream) {
  const float* x    = (const float*)d_in[0];   // [B, NIN]
  const float* slct = (const float*)d_in[1];   // [B, NSLCT]
  const float* W    = (const float*)d_in[2];   // [NSLCT, NHID, NIN]
  const float* bias = (const float*)d_in[3];   // [NSLCT, NHID]
  float* out = (float*)d_out;                  // [B, NHID]
  int* ws = (int*)d_ws;

  init_ws_kernel<<<1, 32, 0, stream>>>(ws);
  argmax_count_kernel<<<B_ROWS / 256, 256, 0, stream>>>(slct, ws);
  scan_offsets_kernel<<<1, 1, 0, stream>>>(ws);
  scatter_rows_kernel<<<B_ROWS / 256, 256, 0, stream>>>(slct, ws);

  dim3 grid(NHID / TN, B_ROWS / TM, NSLCT);
  moe_gemm_kernel<<<grid, THREADS, 0, stream>>>(x, W, bias, ws, out);
}
